// AutoDecoder_82102594831099
// MI455X (gfx1250) — matrix operand order repaired and checked by reading, not compiled, not run on hardware
//
#include <hip/hip_runtime.h>
#include <hip/hip_bf16.h>

// ---------------------------------------------------------------------------
// CDNA5 (gfx1250) types for WMMA
// ---------------------------------------------------------------------------
typedef __attribute__((ext_vector_type(16))) __bf16 v16bf;
typedef __attribute__((ext_vector_type(8)))  __bf16 v8bf;
typedef __attribute__((ext_vector_type(8)))  float  v8f;

#define WMMA_BF16(a, b, c) \
  __builtin_amdgcn_wmma_f32_16x16x32_bf16(false, (a), false, (b), (short)0, (c), false, false)

// Problem constants
#define B_    64
#define LATENT 256
#define HIDDEN 512
#define RC_   10
#define V_    32
#define FD_   32
#define OUTFEAT (3 * V_ * FD_ * RC_)   // 30720
#define EPS_  1e-5f

// ---------------------------------------------------------------------------
// 1) fp32 -> bf16 conversion (z + all weights), 9 segments via blockIdx.y
// ---------------------------------------------------------------------------
struct CvtSegs {
  const float* src[9];
  __bf16*      dst[9];
  int          n[9];
};

__global__ __launch_bounds__(256) void cvt_kernel(CvtSegs s) {
  const int seg = blockIdx.y;
  const int n   = s.n[seg];
  const float* __restrict__ src = s.src[seg];
  __bf16* __restrict__ dst = s.dst[seg];
  for (int i = blockIdx.x * blockDim.x + threadIdx.x; i < n;
       i += gridDim.x * blockDim.x) {
    dst[i] = (__bf16)src[i];
  }
}

// ---------------------------------------------------------------------------
// 2) Fused hidden layer: out = relu(LN(x @ W^T + b  [concat z])) -> bf16
//    4 blocks x 256 threads (8 waves). Block = 16 rows x 512 cols.
// ---------------------------------------------------------------------------
__global__ __launch_bounds__(256) void layer_kernel(
    const __bf16* __restrict__ xin, int in_dim,
    const __bf16* __restrict__ w,   const float* __restrict__ bias,
    int out_gemm,
    const float* __restrict__ zconcat,     // non-null for layer 4
    const float* __restrict__ g, const float* __restrict__ beta,
    __bf16* __restrict__ xout) {

  __shared__ float lds[16][HIDDEN];

  const int m0   = blockIdx.x * 16;
  const int lane = threadIdx.x & 31;
  const int wave = threadIdx.x >> 5;
  const bool lo  = (lane < 16);
  const int il   = lo ? lane : lane - 16;

  // A-operand per-lane addressing (16x32 bf16, contiguous k = 16*(lane/16) + s; B uses the same order)
  const int arow = m0 + il;
  const int ka0  = lo ? 0 : 16;
  const int ka1  = lo ? 8 : 24;
  const int kb0  = lo ? 0 : 16;

  const int ntiles = out_gemm >> 4;
  for (int t = wave; t < ntiles; t += 8) {
    const int n0 = t * 16;
    v8f acc = {};
    const __bf16* wrow = w + (size_t)(n0 + il) * in_dim;
    const __bf16* xrow = xin + (size_t)arow * in_dim;
    for (int kk = 0; kk < in_dim; kk += 32) {
      v8bf alo = *(const v8bf*)(xrow + kk + ka0);
      v8bf ahi = *(const v8bf*)(xrow + kk + ka1);
      v8bf blo = *(const v8bf*)(wrow + kk + kb0);
      v8bf bhi = *(const v8bf*)(wrow + kk + kb0 + 8);
      v16bf a, b;
#pragma unroll
      for (int q = 0; q < 8; q++) {
        a[q] = alo[q]; a[8 + q] = ahi[q];
        b[q] = blo[q]; b[8 + q] = bhi[q];
      }
      acc = WMMA_BF16(a, b, acc);
    }
    const float bn = bias[n0 + il];
#pragma unroll
    for (int rr = 0; rr < 8; rr++) {
      const int M = lo ? rr : 8 + rr;
      lds[M][n0 + il] = acc[rr] + bn;
    }
  }

  // Layer-4 concat: cols [256,512) come from raw z (fp32)
  if (zconcat) {
    for (int t = threadIdx.x; t < 16 * LATENT; t += 256) {
      const int m = t >> 8, c = t & 255;
      lds[m][LATENT + c] = zconcat[(size_t)(m0 + m) * LATENT + c];
    }
  }
  __syncthreads();

  // LayerNorm + ReLU + store bf16. Each wave owns 2 rows.
#pragma unroll
  for (int rr = 0; rr < 2; rr++) {
    const int row = wave * 2 + rr;
    float s = 0.f, s2 = 0.f;
#pragma unroll
    for (int t = 0; t < 16; t++) {
      const float v = lds[row][lane + 32 * t];
      s += v; s2 += v * v;
    }
#pragma unroll
    for (int m = 16; m >= 1; m >>= 1) {
      s  += __shfl_xor(s,  m, 32);
      s2 += __shfl_xor(s2, m, 32);
    }
    const float mu   = s * (1.0f / HIDDEN);
    const float var  = s2 * (1.0f / HIDDEN) - mu * mu;
    const float rstd = rsqrtf(var + EPS_);
#pragma unroll
    for (int t = 0; t < 16; t++) {
      const int n = lane + 32 * t;
      float y = (lds[row][n] - mu) * rstd * g[n] + beta[n];
      y = y > 0.f ? y : 0.f;
      xout[(size_t)(m0 + row) * HIDDEN + n] = (__bf16)y;
    }
  }
}

// ---------------------------------------------------------------------------
// 3) Layer 8: out(64x30720) = x(64x512) @ W8^T + b8   (fp32 out)
//    960 blocks x 8 waves; one 16x16 tile per wave; K=512.
// ---------------------------------------------------------------------------
__global__ __launch_bounds__(256) void layer8_kernel(
    const __bf16* __restrict__ xin, const __bf16* __restrict__ w,
    const float* __restrict__ bias, float* __restrict__ out) {

  const int lane = threadIdx.x & 31;
  const int wave = threadIdx.x >> 5;
  const int id   = blockIdx.x * 8 + wave;    // 0..7679
  const int m0   = (id & 3) * 16;
  const int n0   = (id >> 2) * 16;

  const bool lo = (lane < 16);
  const int il  = lo ? lane : lane - 16;
  const int ka0 = lo ? 0 : 16;
  const int ka1 = lo ? 8 : 24;
  const int kb0 = lo ? 0 : 16;

  v8f acc = {};
  const __bf16* xrow = xin + (size_t)(m0 + il) * HIDDEN;
  const __bf16* wrow = w + (size_t)(n0 + il) * HIDDEN;
#pragma unroll 4
  for (int kk = 0; kk < HIDDEN; kk += 32) {
    v8bf alo = *(const v8bf*)(xrow + kk + ka0);
    v8bf ahi = *(const v8bf*)(xrow + kk + ka1);
    v8bf blo = *(const v8bf*)(wrow + kk + kb0);
    v8bf bhi = *(const v8bf*)(wrow + kk + kb0 + 8);
    v16bf a, b;
#pragma unroll
    for (int q = 0; q < 8; q++) {
      a[q] = alo[q]; a[8 + q] = ahi[q];
      b[q] = blo[q]; b[8 + q] = bhi[q];
    }
    acc = WMMA_BF16(a, b, acc);
  }
  const float bn = bias[n0 + il];
#pragma unroll
  for (int rr = 0; rr < 8; rr++) {
    const int M = lo ? rr : 8 + rr;
    out[(size_t)(m0 + M) * OUTFEAT + n0 + il] = acc[rr] + bn;
  }
}

// ---------------------------------------------------------------------------
// 4) Einsum: grid[b,f,i,j,k] = sum_r X[r,i] Y[r,j] Z[r,k]
//    One block per (b,f) (2048 blocks x 4 waves).
//    Per (b,f): GEMM A(32x10-pad32) x T(10-pad32 x 1024), T[r, j*32+k]=Y*Z.
//    Output is write-once streaming (256 MiB) -> non-temporal stores so x8
//    (7.9 MB, re-read ~32x) stays resident in the 192 MB L2.
// ---------------------------------------------------------------------------
__global__ __launch_bounds__(128) void einsum_kernel(
    const float* __restrict__ x8, float* __restrict__ gout) {

  __shared__ float X[RC_][V_], Y[RC_][V_], Z[RC_][V_];

  const int b = blockIdx.x >> 5;
  const int f = blockIdx.x & 31;

  // Stage the three 10x32 factor matrices (stride-32 gather from x8).
  for (int t = threadIdx.x; t < 3 * RC_ * V_; t += 128) {
    const int s = t / (RC_ * V_);
    const int rem = t - s * (RC_ * V_);
    const int r = rem >> 5, v = rem & 31;
    const float val =
        x8[(size_t)b * OUTFEAT + s * (RC_ * V_ * FD_) + r * (V_ * FD_) + v * FD_ + f];
    (s == 0 ? X : (s == 1 ? Y : Z))[r][v] = val;
  }
  __syncthreads();

  const int lane = threadIdx.x & 31;
  const int wave = threadIdx.x >> 5;
  const bool lo  = (lane < 16);
  const int il   = lo ? lane : lane - 16;

  // A operands for i-tiles 0..15 and 16..31 (K padded 10 -> 32 with zeros).
  v16bf A0, A1;
#pragma unroll
  for (int t = 0; t < 16; t++) {
    const int K = lo ? t : 16 + t;
    const float v0 = (K < RC_) ? X[K][il] : 0.f;
    const float v1 = (K < RC_) ? X[K][16 + il] : 0.f;
    A0[t] = (__bf16)v0;
    A1[t] = (__bf16)v1;
  }

  float* __restrict__ obase = gout + (size_t)blockIdx.x * (V_ * V_ * V_);

  for (int j = wave; j < V_; j += 4) {
#pragma unroll
    for (int k0 = 0; k0 < V_; k0 += 16) {
      // B[K][N] = Y[K][j] * Z[K][k0+N]; lanes>=16 carry K>=16 (all zero-pad).
      v16bf Bv;
#pragma unroll
      for (int t = 0; t < 16; t++) {
        float v = 0.f;
        if (lo && t < RC_) v = Y[t][j] * Z[t][k0 + il];
        Bv[t] = (__bf16)v;
      }
      v8f c0 = {}, c1 = {};
      c0 = WMMA_BF16(A0, Bv, c0);
      c1 = WMMA_BF16(A1, Bv, c1);

      const int k = k0 + il;
#pragma unroll
      for (int rr = 0; rr < 8; rr++) {
        const int M = lo ? rr : 8 + rr;
        __builtin_nontemporal_store(
            c0[rr], &obase[(size_t)M * (V_ * V_) + j * V_ + k]);
        __builtin_nontemporal_store(
            c1[rr], &obase[(size_t)(16 + M) * (V_ * V_) + j * V_ + k]);
      }
    }
  }
}

// ---------------------------------------------------------------------------
// Host launcher
// ---------------------------------------------------------------------------
extern "C" void kernel_launch(void* const* d_in, const int* in_sizes, int n_in,
                              void* d_out, int out_size, void* d_ws, size_t ws_size,
                              hipStream_t stream) {
  (void)in_sizes; (void)n_in; (void)out_size; (void)ws_size;

  const float* z = (const float*)d_in[0];
  const float* W[9]; const float* Bi[9]; const float* G[8]; const float* Be[8];
  int idx = 1;
  for (int i = 1; i <= 7; i++) {
    W[i]  = (const float*)d_in[idx++];
    Bi[i] = (const float*)d_in[idx++];
    G[i]  = (const float*)d_in[idx++];
    Be[i] = (const float*)d_in[idx++];
  }
  W[8]  = (const float*)d_in[idx++];
  Bi[8] = (const float*)d_in[idx++];

  auto alignup = [](size_t x) { return (x + 255) & ~size_t(255); };
  static const int wsz[9] = {0, HIDDEN * LATENT, HIDDEN * HIDDEN, HIDDEN * HIDDEN,
                             LATENT * HIDDEN, HIDDEN * HIDDEN, HIDDEN * HIDDEN,
                             HIDDEN * HIDDEN, OUTFEAT * HIDDEN};

  char* p = (char*)d_ws;
  __bf16* zb = (__bf16*)p; p += alignup((size_t)B_ * LATENT * sizeof(__bf16));
  __bf16* Wb[9];
  for (int i = 1; i <= 8; i++) { Wb[i] = (__bf16*)p; p += alignup((size_t)wsz[i] * sizeof(__bf16)); }
  __bf16* actA = (__bf16*)p; p += alignup((size_t)B_ * HIDDEN * sizeof(__bf16));
  __bf16* actB = (__bf16*)p; p += alignup((size_t)B_ * HIDDEN * sizeof(__bf16));
  float*  x8   = (float*)p;  p += alignup((size_t)B_ * OUTFEAT * sizeof(float));

  // 1) convert z + weights to bf16
  CvtSegs segs;
  segs.src[0] = z; segs.dst[0] = zb; segs.n[0] = B_ * LATENT;
  for (int i = 1; i <= 8; i++) { segs.src[i] = W[i]; segs.dst[i] = Wb[i]; segs.n[i] = wsz[i]; }
  cvt_kernel<<<dim3(1024, 9), 256, 0, stream>>>(segs);

  // 2) hidden layers 1..7 (ping-pong bf16 activations)
  const __bf16* xin = zb;
  __bf16* xout = actA;
  int in_dim = LATENT;
  for (int i = 1; i <= 7; i++) {
    const int out_gemm = (i == 4) ? (HIDDEN - LATENT) : HIDDEN;
    const float* zc = (i == 4) ? z : nullptr;
    layer_kernel<<<4, 256, 0, stream>>>(xin, in_dim, Wb[i], Bi[i], out_gemm, zc,
                                        G[i], Be[i], xout);
    xin = xout;
    xout = (xout == actA) ? actB : actA;
    in_dim = HIDDEN;
  }

  // 3) layer 8 -> x8 (fp32, 64 x 30720)
  layer8_kernel<<<960, 256, 0, stream>>>(xin, Wb[8], Bi[8], x8);

  // 4) triple outer-product einsum -> 256 MiB output (NT streaming stores)
  einsum_kernel<<<2048, 128, 0, stream>>>(x8, (float*)d_out);
}
